// LiteFFBlock_58274116272318
// MI455X (gfx1250) — matrix operand order repaired and checked by reading, not compiled, not run on hardware
//
#include <hip/hip_runtime.h>
#include <hip/hip_bf16.h>

// ---------------------------------------------------------------------------
// Types for CDNA5 WMMA (wave32): v_wmma_f32_16x16x32_bf16
// ---------------------------------------------------------------------------
typedef __attribute__((ext_vector_type(16))) __bf16 v16bf;
typedef __attribute__((ext_vector_type(8)))  float  v8f;

struct U4x2 { uint4 a, b; };

static __device__ __forceinline__ v16bf make_frag(uint4 lo, uint4 hi) {
    U4x2 t; t.a = lo; t.b = hi;
    return __builtin_bit_cast(v16bf, t);
}

static __device__ __forceinline__ unsigned short f2bf(float x) {
    unsigned u = __builtin_bit_cast(unsigned, x);
    u += 0x7FFFu + ((u >> 16) & 1u);           // round-to-nearest-even
    return (unsigned short)(u >> 16);
}

// generic -> LDS(AS3) byte offset, for async-to-LDS destination operands
static __device__ __forceinline__ unsigned to_lds(const void* p) {
    return (unsigned)(unsigned long long)(__attribute__((address_space(3))) const void*)p;
}

// one wave-wide async 16B/lane copy; IOFFSET applies to BOTH LDS dst and
// global src (see ISA 08_async_tensor)
#define ASYNC_B128(ldsoff, gaddr, imm)                                         \
    asm volatile("global_load_async_to_lds_b128 %0, %1, off offset:%c2"        \
                 :: "v"(ldsoff), "v"(gaddr), "n"(imm) : "memory")

// ---------------------------------------------------------------------------
// Elementwise helpers
// ---------------------------------------------------------------------------
__global__ __launch_bounds__(256) void cvt_f32_bf16_kernel(
    const float* __restrict__ src, unsigned short* __restrict__ dst, int n) {
    int i = blockIdx.x * 256 + threadIdx.x;
    if (i < n) dst[i] = f2bf(src[i]);
}

// f32 [R][C] -> bf16 [C][R]   (tiled, coalesced both ways)
__global__ __launch_bounds__(256) void cvt_transpose_kernel(
    const float* __restrict__ src, unsigned short* __restrict__ dst, int R, int C) {
    __shared__ unsigned short tile[32][33];
    int c0 = blockIdx.x * 32, r0 = blockIdx.y * 32;
    int tx = threadIdx.x & 31, ty = threadIdx.x >> 5;   // 8 rows of 32
#pragma unroll
    for (int i = ty; i < 32; i += 8)
        tile[i][tx] = f2bf(src[(long)(r0 + i) * C + c0 + tx]);
    __syncthreads();
#pragma unroll
    for (int i = ty; i < 32; i += 8)
        dst[(long)(c0 + i) * R + r0 + tx] = tile[tx][i];
}

// avg-pool kernel=2 stride=2 over time for q[batch 3] -> bf16 [2048,512]
__global__ __launch_bounds__(256) void avgpool_q3_kernel(
    const float* __restrict__ q3, unsigned short* __restrict__ qp3, int dk) {
    int idx = blockIdx.x * 256 + threadIdx.x;
    int row = idx / dk, d = idx % dk;
    float a = q3[(2 * row) * dk + d];
    float b = q3[(2 * row + 1) * dk + d];
    qp3[idx] = f2bf(0.5f * (a + b));
}

// softmax(floor(scores / sqrt(D_V))) over a 2048-wide row -> bf16
__global__ __launch_bounds__(256) void softmax_floor_kernel(
    const float* __restrict__ S, unsigned short* __restrict__ A, int n) {
    __shared__ float red[256];
    const float inv = 0.02209708691207961f;             // 1/sqrt(2048)
    long row = blockIdx.x;
    const float* s = S + row * n;
    float m = -3.0e38f;
    for (int i = threadIdx.x; i < n; i += 256) m = fmaxf(m, floorf(s[i] * inv));
    red[threadIdx.x] = m; __syncthreads();
    for (int st = 128; st > 0; st >>= 1) {
        if (threadIdx.x < st) red[threadIdx.x] = fmaxf(red[threadIdx.x], red[threadIdx.x + st]);
        __syncthreads();
    }
    m = red[0]; __syncthreads();
    float sum = 0.f;
    for (int i = threadIdx.x; i < n; i += 256) sum += __expf(floorf(s[i] * inv) - m);
    red[threadIdx.x] = sum; __syncthreads();
    for (int st = 128; st > 0; st >>= 1) {
        if (threadIdx.x < st) red[threadIdx.x] += red[threadIdx.x + st];
        __syncthreads();
    }
    float rs = 1.f / red[0];
    for (int i = threadIdx.x; i < n; i += 256)
        A[row * n + i] = f2bf(__expf(floorf(s[i] * inv) - m) * rs);
}

// LayerNorm (biased variance, eps=1e-6) over 2048-wide rows -> bf16
__global__ __launch_bounds__(256) void layernorm_kernel(
    const float* __restrict__ X, const float* __restrict__ g,
    const float* __restrict__ be, unsigned short* __restrict__ Y, int n) {
    __shared__ float rs_[256];
    __shared__ float rq_[256];
    long row = blockIdx.x;
    const float* x = X + row * n;
    float s = 0.f, s2 = 0.f;
    for (int i = threadIdx.x; i < n; i += 256) { float v = x[i]; s += v; s2 += v * v; }
    rs_[threadIdx.x] = s; rq_[threadIdx.x] = s2; __syncthreads();
    for (int st = 128; st > 0; st >>= 1) {
        if (threadIdx.x < st) { rs_[threadIdx.x] += rs_[threadIdx.x + st];
                                rq_[threadIdx.x] += rq_[threadIdx.x + st]; }
        __syncthreads();
    }
    float mu   = rs_[0] / n;
    float var  = rq_[0] / n - mu * mu;
    float rstd = rsqrtf(var + 1e-6f);
    for (int i = threadIdx.x; i < n; i += 256)
        Y[row * n + i] = f2bf((x[i] - mu) * rstd * g[i] + be[i]);
}

// ---------------------------------------------------------------------------
// Tiled bf16 WMMA GEMM: C = act(A @ Bt^T + bias + residual)
//   A  : [M,K] bf16 row-major.  Bt : [N,K] bf16 row-major (pre-transposed).
//   Block 256 thr = 8 waves; tile 128x128, stage depth BK=64 (2 wmma K-steps).
//   Double-buffered LDS filled with GLOBAL_LOAD_ASYNC_TO_LDS_B128 (ASYNCcnt),
//   one s_wait_asynccnt + one barrier per 64-deep stage.
//   Wave (wm 0..3, wn 0..1) owns 32x64 -> 16 v_wmma_f32_16x16x32_bf16 / stage.
//   OUTMODE: 0 = f32, 1 = bf16.
// ---------------------------------------------------------------------------
template <int OUTMODE>
__global__ __launch_bounds__(256) void gemm_bf16_kernel(
    const unsigned short* __restrict__ A, long aStride,
    const unsigned short* __restrict__ Bt, long bStride,
    void* __restrict__ Cvoid, long cStride,
    const float* __restrict__ bias,
    const float* __restrict__ residual, long rStride,
    int K, int lda, int ldbt, int ldc, int doRelu) {

    __shared__ __align__(16) unsigned short sA[2][128][64];   // [buf][m][k] 32KB
    __shared__ __align__(16) unsigned short sB[2][128][64];   // [buf][n][k] 32KB

    const int tid  = threadIdx.x;
    const int lane = tid & 31;
    const int wave = tid >> 5;
    const int wm   = wave & 3;
    const int wn   = wave >> 2;
    const int m0   = blockIdx.y * 128;
    const int n0   = blockIdx.x * 128;
    const int bz   = blockIdx.z;

    A  += (long)bz * aStride;
    Bt += (long)bz * bStride;

    const int lrow  = lane & 15;
    const int khalf = lane >> 4;

    // per-thread async staging: row = tid/2, 32-wide K half = tid&1, 4 x b128
    // each; the four chunks share one base address and use offset: immediates
    const int srow  = tid >> 1;
    const int shalf = (tid & 1) * 32;

    auto issue_stage = [&](int buf, int k0) {
        const unsigned short* ga = A  + (long)(m0 + srow) * lda  + k0 + shalf;
        const unsigned short* gb = Bt + (long)(n0 + srow) * ldbt + k0 + shalf;
        unsigned la = to_lds(&sA[buf][srow][shalf]);
        unsigned lb = to_lds(&sB[buf][srow][shalf]);
        ASYNC_B128(la, ga, 0);
        ASYNC_B128(la, ga, 16);
        ASYNC_B128(la, ga, 32);
        ASYNC_B128(la, ga, 48);
        ASYNC_B128(lb, gb, 0);
        ASYNC_B128(lb, gb, 16);
        ASYNC_B128(lb, gb, 32);
        ASYNC_B128(lb, gb, 48);
    };

    v8f acc[2][4];
#pragma unroll
    for (int i = 0; i < 2; ++i)
#pragma unroll
        for (int j = 0; j < 4; ++j) acc[i][j] = (v8f){};

    const int nk = K >> 6;                     // stages of 64
    issue_stage(0, 0);
    asm volatile("s_wait_asynccnt 0x0" ::: "memory");
    __syncthreads();

    for (int ks = 0; ks < nk; ++ks) {
        const int cur = ks & 1;
        if (ks + 1 < nk) issue_stage(cur ^ 1, (ks + 1) * 64);

#pragma unroll
        for (int kk = 0; kk < 64; kk += 32) {
            v16bf af[2], bfr[4];
#pragma unroll
            for (int i = 0; i < 2; ++i) {
                const uint4* p = (const uint4*)&sA[cur][wm * 32 + i * 16 + lrow][kk];
                af[i] = make_frag(p[khalf], p[2 + khalf]);        // K {0-7,16-23}/{8-15,24-31}
            }
#pragma unroll
            for (int j = 0; j < 4; ++j) {
                const uint4* p = (const uint4*)&sB[cur][wn * 64 + j * 16 + lrow][kk];
                bfr[j] = make_frag(p[khalf], p[2 + khalf]);        // K {0-7,16-23}/{8-15,24-31}
            }
#pragma unroll
            for (int i = 0; i < 2; ++i)
#pragma unroll
                for (int j = 0; j < 4; ++j)
                    acc[i][j] = __builtin_amdgcn_wmma_f32_16x16x32_bf16(
                        false, af[i], false, bfr[j], (short)0, acc[i][j], false, false);
        }
        asm volatile("s_wait_asynccnt 0x0" ::: "memory");
        __syncthreads();
    }

    // ---- epilogue: C layout VGPR r -> row = r + 8*(lane/16), col = lane%16
    float* Cf          = (float*)Cvoid;
    unsigned short* Cb = (unsigned short*)Cvoid;
    long cOff = (long)bz * cStride;
    long rOff = (long)bz * rStride;
#pragma unroll
    for (int i = 0; i < 2; ++i) {
#pragma unroll
        for (int j = 0; j < 4; ++j) {
            int col     = n0 + wn * 64 + j * 16 + lrow;
            int rowBase = m0 + wm * 32 + i * 16 + (khalf << 3);
            float bv_ = bias ? bias[col] : 0.f;
#pragma unroll
            for (int r = 0; r < 8; ++r) {
                int row = rowBase + r;
                float val = acc[i][j][r] + bv_;
                if (residual) val += residual[rOff + (long)row * ldc + col];
                if (doRelu)   val = val > 0.f ? val : 0.f;
                if (OUTMODE == 0) Cf[cOff + (long)row * ldc + col] = val;
                else              Cb[cOff + (long)row * ldc + col] = f2bf(val);
            }
        }
    }
}

// ---------------------------------------------------------------------------
// Host orchestration
// ---------------------------------------------------------------------------
extern "C" void kernel_launch(void* const* d_in, const int* in_sizes, int n_in,
                              void* d_out, int out_size, void* d_ws, size_t ws_size,
                              hipStream_t stream) {
    const float* q    = (const float*)d_in[0];
    const float* k    = (const float*)d_in[1];
    const float* v    = (const float*)d_in[2];
    const float* Wq   = (const float*)d_in[3];
    const float* bq   = (const float*)d_in[4];
    const float* Wk   = (const float*)d_in[5];
    const float* bk   = (const float*)d_in[6];
    const float* Wv   = (const float*)d_in[7];
    const float* bv   = (const float*)d_in[8];
    const float* ln_g = (const float*)d_in[9];
    const float* ln_b = (const float*)d_in[10];
    const float* W1   = (const float*)d_in[11];
    const float* b1   = (const float*)d_in[12];
    const float* W2   = (const float*)d_in[13];
    const float* b2   = (const float*)d_in[14];
    float* out = (float*)d_out;

    constexpr int BS = 4, LQ = 4096, LKV = 2048, DK = 512, DV = 2048, DH = 8192;
    constexpr int M = LQ / 2;                 // 2048 pooled query positions

    size_t off = 0;
    auto wsa = [&](size_t bytes) -> void* {
        void* p = (void*)((char*)d_ws + off);
        off = (off + bytes + 255) & ~(size_t)255;
        return p;
    };
    unsigned short* qp3_bf = (unsigned short*)wsa((size_t)M * DK * 2);
    unsigned short* k3_bf  = (unsigned short*)wsa((size_t)LKV * DK * 2);
    unsigned short* WqT    = (unsigned short*)wsa((size_t)DK * DK * 2);
    unsigned short* WkT    = (unsigned short*)wsa((size_t)DK * DK * 2);
    unsigned short* WvT    = (unsigned short*)wsa((size_t)DV * DV * 2);
    unsigned short* v_bf   = (unsigned short*)wsa((size_t)BS * LKV * DV * 2);
    unsigned short* W1T    = (unsigned short*)wsa((size_t)DV * DH * 2);
    unsigned short* W2T    = (unsigned short*)wsa((size_t)DH * DV * 2);
    unsigned short* qh3_bf = (unsigned short*)wsa((size_t)M * DK * 2);
    unsigned short* kh3_bf = (unsigned short*)wsa((size_t)LKV * DK * 2);
    float*          scores = (float*)wsa((size_t)M * LKV * 4);
    unsigned short* A_bf   = (unsigned short*)wsa((size_t)M * LKV * 2);
    unsigned short* vh_bf  = (unsigned short*)wsa((size_t)BS * LKV * DV * 2);
    float*          attn   = (float*)wsa((size_t)BS * M * LKV * 4);
    unsigned short* ln_bf  = (unsigned short*)wsa((size_t)BS * M * DV * 2);
    unsigned short* h_bf   = (unsigned short*)wsa((size_t)BS * M * DH * 2);

    dim3 blk(256);

    // ---- one-time conversions; weights go to [N,K] bf16 so every GEMM's B
    //      operand is contiguous (async b128-friendly). Only batch 3 of q/k
    //      feeds scores[-1].
    avgpool_q3_kernel<<<(M * DK) / 256, blk, 0, stream>>>(q + (size_t)3 * LQ * DK, qp3_bf, DK);
    cvt_f32_bf16_kernel<<<(LKV * DK) / 256, blk, 0, stream>>>(k + (size_t)3 * LKV * DK, k3_bf, LKV * DK);
    cvt_f32_bf16_kernel<<<(BS * LKV * DV) / 256, blk, 0, stream>>>(v, v_bf, BS * LKV * DV);
    cvt_transpose_kernel<<<dim3(DK / 32, DK / 32), blk, 0, stream>>>(Wq, WqT, DK, DK);
    cvt_transpose_kernel<<<dim3(DK / 32, DK / 32), blk, 0, stream>>>(Wk, WkT, DK, DK);
    cvt_transpose_kernel<<<dim3(DV / 32, DV / 32), blk, 0, stream>>>(Wv, WvT, DV, DV);
    cvt_transpose_kernel<<<dim3(DH / 32, DV / 32), blk, 0, stream>>>(W1, W1T, DV, DH);
    cvt_transpose_kernel<<<dim3(DV / 32, DH / 32), blk, 0, stream>>>(W2, W2T, DH, DV);

    // qh3 = qp3 @ Wq + bq            [2048,512]  -> bf16
    gemm_bf16_kernel<1><<<dim3(DK / 128, M / 128, 1), blk, 0, stream>>>(
        qp3_bf, 0, WqT, 0, qh3_bf, 0, bq, nullptr, 0, DK, DK, DK, DK, 0);
    // kh3 = k3 @ Wk + bk             [2048,512]  -> bf16
    gemm_bf16_kernel<1><<<dim3(DK / 128, LKV / 128, 1), blk, 0, stream>>>(
        k3_bf, 0, WkT, 0, kh3_bf, 0, bk, nullptr, 0, DK, DK, DK, DK, 0);
    // vh[b] = v[b] @ Wv + bv         [2048,2048] -> bf16 (row-major == Bt for attn)
    gemm_bf16_kernel<1><<<dim3(DV / 128, LKV / 128, BS), blk, 0, stream>>>(
        v_bf, (long)LKV * DV, WvT, 0, vh_bf, (long)LKV * DV, bv, nullptr, 0,
        DV, DV, DV, DV, 0);
    // scores = qh3 @ kh3^T           [2048,2048] -> f32   (kh3 is already [N,K])
    gemm_bf16_kernel<0><<<dim3(LKV / 128, M / 128, 1), blk, 0, stream>>>(
        qh3_bf, 0, kh3_bf, 0, scores, 0, nullptr, nullptr, 0, DK, DK, DK, LKV, 0);
    // A = softmax(floor(scores / sqrt(DV)))
    softmax_floor_kernel<<<M, blk, 0, stream>>>(scores, A_bf, LKV);
    // attn[b] = A @ vh[b]^T + v[b]   [2048,2048] -> f32 (residual fused)
    gemm_bf16_kernel<0><<<dim3(LKV / 128, M / 128, BS), blk, 0, stream>>>(
        A_bf, 0, vh_bf, (long)LKV * DV, attn, (long)M * LKV,
        nullptr, v, (long)LKV * DV, DV, LKV, DV, LKV, 0);
    // LayerNorm -> bf16
    layernorm_kernel<<<BS * M, blk, 0, stream>>>(attn, ln_g, ln_b, ln_bf, DV);
    // h[b] = relu(ln[b] @ W1 + b1)   [2048,8192] -> bf16 (relu fused)
    gemm_bf16_kernel<1><<<dim3(DH / 128, M / 128, BS), blk, 0, stream>>>(
        ln_bf, (long)M * DV, W1T, 0, h_bf, (long)M * DH, b1, nullptr, 0,
        DV, DV, DV, DH, 1);
    // out[b] = h[b] @ W2 + b2        [2048,2048] -> f32 to d_out
    gemm_bf16_kernel<0><<<dim3(DV / 128, M / 128, BS), blk, 0, stream>>>(
        h_bf, (long)M * DH, W2T, 0, out, (long)M * DV, b2, nullptr, 0,
        DH, DH, DH, DV, 0);
}
